// pic_transformer_26388279066649
// MI455X (gfx1250) — matrix operand order repaired and checked by reading, not compiled, not run on hardware
//
#include <hip/hip_runtime.h>
#include <hip/hip_bf16.h>
#include <math.h>
#include <stdint.h>

// ---------------- problem constants ----------------
#define BSZ   32
#define NPIC  4
#define CCH   3
#define HH    224
#define PP    16
#define HP    14          // 224/16
#define NPAT  196         // 14*14
#define DDIM  768         // 3*16*16
#define MROWS 25088       // 32*4*196
#define SP    224         // padded sequence (14*16, 7*32)
#define NSEG  128         // 32*4
#define WELEM (768*768)

typedef __attribute__((ext_vector_type(16))) _Float16 v16h;
typedef __attribute__((ext_vector_type(8)))  _Float16 v8h;
typedef __attribute__((ext_vector_type(8)))  float    v8f;

// global (device memory) address space — forces global_load/global_store
#define AS1 __attribute__((address_space(1)))
typedef const _Float16 AS1* gcf16;
typedef       _Float16 AS1* gf16;
typedef const v8h      AS1* gcv8h;
typedef const float    AS1* gcf32;
typedef       float    AS1* gf32;

__device__ __forceinline__ gcf16 as_g(const _Float16* p) { return (gcf16)(uintptr_t)p; }
__device__ __forceinline__ gf16  as_g(_Float16* p)       { return (gf16)(uintptr_t)p;  }
__device__ __forceinline__ gcf32 as_g(const float* p)    { return (gcf32)(uintptr_t)p; }
__device__ __forceinline__ gf32  as_g(float* p)          { return (gf32)(uintptr_t)p;  }

union FragU { v16h v; v8h h[2]; };

#define FLAG_RELU  1
#define FLAG_F32   2
#define FLAG_TRANS 4

__device__ __forceinline__ int iclamp(int v, int hi) { return v < hi ? v : hi; }

// ---------------- generic segmented WMMA GEMM ----------------
// C[seg] = act( alpha * A[seg] @ Bt[seg]^T + bias )
// 256 threads = 8 waves in 4(M) x 2(N); each wave computes a 32x64 tile as
// 2x4 of 16x16 WMMA tiles -> macro tile 128x128.  Fragment row/col indices
// are clamped (not masked) BEFORE the K loop, so the inner loop is
// branch-free: 12x global_load_b128 + 8x v_wmma per K=32 step.
//
// A fragment (16x32 f16, ISA 7.12.2): lane l: m=l&15, g=l>>4;
//   halves 0..7 -> K=g*8+e ; halves 8..15 -> K=16+g*8+e  (two 16B loads)
// B fragment from Bt[N][K]: lane l: n=l&15, g=l>>4;
//   halves 0..7 -> K=g*8+e ; halves 8..15 -> K=16+g*8+e  (two 16B loads)
__global__ __launch_bounds__(256)
void wmma_gemm_seg(const _Float16* __restrict__ A, long lda, long aSegStride,
                   int aSegDiv, int aValid,
                   const _Float16* __restrict__ Bt, long ldbt, long bSegStride,
                   int bValid,
                   const float* __restrict__ bias,
                   void* __restrict__ Cv, long ldc, long cSegStride,
                   int cRows, int Ncols, int Kdim, float alpha, int flags) {
    int seg  = blockIdx.z;
    gcf16 Aseg = as_g(A) + (long)(seg / aSegDiv) * aSegStride;
    gcf16 Bseg = as_g(Bt) + (long)seg * bSegStride;

    int lane = threadIdx.x & 31;
    int wave = threadIdx.x >> 5;
    int wm = wave & 3;       // 0..3  (M)
    int wn = wave >> 2;      // 0..1  (N)
    int m0 = blockIdx.x * 128 + wm * 32;
    int n0 = blockIdx.y * 128 + wn * 64;

    int g  = lane >> 4;      // half-wave group
    int hl = lane & 15;

    // clamped per-lane base pointers (loop-invariant, always in-bounds)
    gcf16 pa[2];
    #pragma unroll
    for (int i = 0; i < 2; ++i) {
        int row = iclamp(m0 + i * 16 + hl, aValid - 1);
        pa[i] = Aseg + (long)row * lda + g * 8;
    }
    gcf16 pb[4];
    #pragma unroll
    for (int j = 0; j < 4; ++j) {
        int col = iclamp(n0 + j * 16 + hl, bValid - 1);
        pb[j] = Bseg + (long)col * ldbt + g * 8;
    }

    v8f acc[2][4];
    #pragma unroll
    for (int i = 0; i < 2; ++i)
        #pragma unroll
        for (int j = 0; j < 4; ++j)
            #pragma unroll
            for (int r = 0; r < 8; ++r) acc[i][j][r] = 0.0f;

    for (int kk = 0; kk < Kdim; kk += 32) {
        FragU a[2], b[4];
        #pragma unroll
        for (int i = 0; i < 2; ++i) {
            a[i].h[0] = *(gcv8h)(pa[i]);
            a[i].h[1] = *(gcv8h)(pa[i] + 16);
            pa[i] += 32;
        }
        #pragma unroll
        for (int j = 0; j < 4; ++j) {
            b[j].h[0] = *(gcv8h)(pb[j]);
            b[j].h[1] = *(gcv8h)(pb[j] + 16);
            pb[j] += 32;
        }
        #pragma unroll
        for (int i = 0; i < 2; ++i)
            #pragma unroll
            for (int j = 0; j < 4; ++j)
                acc[i][j] = __builtin_amdgcn_wmma_f32_16x16x32_f16(
                    false, a[i].v, false, b[j].v, (short)0, acc[i][j],
                    false, false);
    }

    // epilogue: C/D layout -> element r: row = r + 8*g, col = lane&15
    gf16  c16   = (gf16)(uintptr_t)Cv;
    gf32  c32   = (gf32)(uintptr_t)Cv;
    gcf32 biasg = (gcf32)(uintptr_t)bias;
    #pragma unroll
    for (int im = 0; im < 2; ++im) {
        #pragma unroll
        for (int in = 0; in < 4; ++in) {
            #pragma unroll
            for (int r = 0; r < 8; ++r) {
                int row = m0 + im * 16 + g * 8 + r;
                int col = n0 + in * 16 + hl;
                if (row >= cRows || col >= Ncols) continue;
                float v = acc[im][in][r] * alpha;
                if (bias) v += biasg[col];
                if (flags & FLAG_RELU) v = fmaxf(v, 0.0f);
                long base = (long)seg * cSegStride;
                if (flags & FLAG_TRANS) {
                    c16[base + (long)col * ldc + row] = (_Float16)v;
                } else if (flags & FLAG_F32) {
                    c32[base + (long)row * ldc + col] = v;
                } else {
                    c16[base + (long)row * ldc + col] = (_Float16)v;
                }
            }
        }
    }
}

// ---------------- weight convert + transpose: Wt[n][k] = (f16) W[k][n] ----------------
__global__ __launch_bounds__(256)
void convert_weights(const float* __restrict__ w1, const float* __restrict__ wq,
                     const float* __restrict__ wk, const float* __restrict__ wv,
                     const float* __restrict__ wo1, const float* __restrict__ wo2,
                     _Float16* __restrict__ out) {
    long id = (long)blockIdx.x * blockDim.x + threadIdx.x;
    if (id >= 6L * WELEM) return;
    int w = (int)(id / WELEM);
    int rem = (int)(id % WELEM);
    int n = rem / DDIM;
    int k = rem % DDIM;
    const float* src;
    switch (w) {
        case 0:  src = w1;  break;
        case 1:  src = wq;  break;
        case 2:  src = wk;  break;
        case 3:  src = wv;  break;
        case 4:  src = wo1; break;
        default: src = wo2; break;
    }
    as_g(out)[(long)w * WELEM + (long)n * DDIM + k] =
        (_Float16)as_g(src)[(long)k * DDIM + n];
}

// ---------------- patchify: x[b,n,c,h,w] -> P[(b*4+n)*196+s][c*256+py*16+px] ----------------
__global__ __launch_bounds__(256)
void patchify(const float* __restrict__ x, _Float16* __restrict__ P) {
    long id = (long)blockIdx.x * blockDim.x + threadIdx.x;
    if (id >= (long)MROWS * DDIM) return;
    int row = (int)(id / DDIM);
    int d   = (int)(id % DDIM);
    int b   = row / (NPIC * NPAT);
    int rem = row % (NPIC * NPAT);
    int n   = rem / NPAT;
    int s   = rem % NPAT;
    int hp  = s / HP, wp = s % HP;
    int c   = d / (PP * PP);
    int rr  = d % (PP * PP);
    int py  = rr / PP, px = rr % PP;
    long xi = ((((long)(b * NPIC + n) * CCH + c) * HH + hp * PP + py) * HH) + wp * PP + px;
    as_g(P)[id] = (_Float16)as_g(x)[xi];
}

// ---------------- masked softmax over 196 valid keys, write 224 padded f16 ----------------
__global__ __launch_bounds__(256)
void softmax_rows(const float* __restrict__ scores, _Float16* __restrict__ attn) {
    int id = blockIdx.x * blockDim.x + threadIdx.x;
    if (id >= NSEG * NPAT) return;
    int seg = id / NPAT;
    int s   = id % NPAT;
    gcf32 sr = as_g(scores) + ((long)seg * SP + s) * SP;
    gf16  ar = as_g(attn)   + ((long)seg * SP + s) * SP;
    float m = -3.4e38f;
    for (int j = 0; j < NPAT; ++j) m = fmaxf(m, sr[j]);
    float sum = 0.0f;
    for (int j = 0; j < NPAT; ++j) sum += __expf(sr[j] - m);
    float inv = 1.0f / sum;
    for (int j = 0; j < NPAT; ++j) ar[j] = (_Float16)(__expf(sr[j] - m) * inv);
    for (int j = NPAT; j < SP; ++j) ar[j] = (_Float16)0.0f;
}

// ---------------- host orchestration ----------------
extern "C" void kernel_launch(void* const* d_in, const int* in_sizes, int n_in,
                              void* d_out, int out_size, void* d_ws, size_t ws_size,
                              hipStream_t stream) {
    const float* x   = (const float*)d_in[0];
    const float* W1  = (const float*)d_in[1];   const float* b1  = (const float*)d_in[2];
    const float* Wq  = (const float*)d_in[3];   const float* bq  = (const float*)d_in[4];
    const float* Wk  = (const float*)d_in[5];   const float* bk  = (const float*)d_in[6];
    const float* Wv  = (const float*)d_in[7];   const float* bv  = (const float*)d_in[8];
    const float* Wo1 = (const float*)d_in[9];   const float* bo1 = (const float*)d_in[10];
    const float* Wo2 = (const float*)d_in[11];  const float* bo2 = (const float*)d_in[12];

    // workspace carve-out (256B aligned)
    char* ws = (char*)d_ws;
    size_t off = 0;
    auto alloc = [&](size_t bytes) -> char* {
        char* p = ws + off;
        off += (bytes + 255) & ~(size_t)255;
        return p;
    };
    _Float16* wt  = (_Float16*)alloc(6L * WELEM * 2);                // transposed f16 weights
    _Float16* p16 = (_Float16*)alloc((size_t)MROWS * DDIM * 2);      // patchified; reused as WV
    _Float16* h16 = (_Float16*)alloc((size_t)MROWS * DDIM * 2);      // hidden; reused as R
    _Float16* q16 = (_Float16*)alloc((size_t)BSZ * SP * DDIM * 2);
    _Float16* k16 = (_Float16*)alloc((size_t)NSEG * SP * DDIM * 2);
    _Float16* vt16= (_Float16*)alloc((size_t)NSEG * DDIM * SP * 2);  // V transposed [d][s]
    float*    sc32= (float*)   alloc((size_t)NSEG * SP * SP * 4);
    _Float16* at16= (_Float16*)alloc((size_t)NSEG * SP * SP * 2);
    _Float16* wv16 = p16;   // P dead after H
    _Float16* r16  = h16;   // H dead after QKV
    (void)ws_size;

    const float scale = 1.0f / sqrtf((float)DDIM);

    // 1) weight convert+transpose
    {
        long n = 6L * WELEM;
        convert_weights<<<dim3((unsigned)((n + 255) / 256)), dim3(256), 0, stream>>>(
            W1, Wq, Wk, Wv, Wo1, Wo2, wt);
    }
    // 2) patchify
    {
        long n = (long)MROWS * DDIM;
        patchify<<<dim3((unsigned)((n + 255) / 256)), dim3(256), 0, stream>>>(x, p16);
    }
    // 3) H = P @ W1t + b1   [25088 x 768]
    wmma_gemm_seg<<<dim3(MROWS / 128, DDIM / 128, 1), dim3(256), 0, stream>>>(
        p16, DDIM, 0L, 1, MROWS,
        wt + 0L * WELEM, DDIM, 0L, DDIM,
        b1, h16, DDIM, 0L, MROWS, DDIM, DDIM, 1.0f, 0);
    // 4) Q = H[:,0] @ Wqt + bq  -> [32 x 224pad x 768]
    wmma_gemm_seg<<<dim3(2, DDIM / 128, BSZ), dim3(256), 0, stream>>>(
        h16, DDIM, (long)NPIC * NPAT * DDIM, 1, NPAT,
        wt + 1L * WELEM, DDIM, 0L, DDIM,
        bq, q16, DDIM, (long)SP * DDIM, SP, DDIM, DDIM, 1.0f, 0);
    // 5) K = H @ Wkt + bk  -> [128 x 224pad x 768]
    wmma_gemm_seg<<<dim3(2, DDIM / 128, NSEG), dim3(256), 0, stream>>>(
        h16, DDIM, (long)NPAT * DDIM, 1, NPAT,
        wt + 2L * WELEM, DDIM, 0L, DDIM,
        bk, k16, DDIM, (long)SP * DDIM, SP, DDIM, DDIM, 1.0f, 0);
    // 6) Vt = (H @ Wvt + bv)^T  -> [128 x 768 x 224pad]
    wmma_gemm_seg<<<dim3(2, DDIM / 128, NSEG), dim3(256), 0, stream>>>(
        h16, DDIM, (long)NPAT * DDIM, 1, NPAT,
        wt + 3L * WELEM, DDIM, 0L, DDIM,
        bv, vt16, SP, (long)DDIM * SP, SP, DDIM, DDIM, 1.0f, FLAG_TRANS);
    // 7) scores = scale * Q @ K^T  -> f32 [128 x 224 x 224]  (K row-major == Bt layout)
    wmma_gemm_seg<<<dim3(2, (SP + 127) / 128, NSEG), dim3(256), 0, stream>>>(
        q16, DDIM, (long)SP * DDIM, NPIC, NPAT,
        k16, DDIM, (long)SP * DDIM, NPAT,
        nullptr, sc32, SP, (long)SP * SP, SP, SP, DDIM, scale, FLAG_F32);
    // 8) masked softmax -> attn f16 [128 x 224 x 224], cols >=196 zeroed
    softmax_rows<<<dim3((NSEG * NPAT + 255) / 256), dim3(256), 0, stream>>>(sc32, at16);
    // 9) WV = attn @ V  (B from Vt, contiguous)  -> compact [25088 x 768]
    wmma_gemm_seg<<<dim3(2, DDIM / 128, NSEG), dim3(256), 0, stream>>>(
        at16, SP, (long)SP * SP, 1, NPAT,
        vt16, SP, (long)DDIM * SP, DDIM,
        nullptr, wv16, DDIM, (long)NPAT * DDIM, NPAT, DDIM, SP, 1.0f, 0);
    // 10) R = relu(WV @ Wo1t + bo1)
    wmma_gemm_seg<<<dim3(MROWS / 128, DDIM / 128, 1), dim3(256), 0, stream>>>(
        wv16, DDIM, 0L, 1, MROWS,
        wt + 4L * WELEM, DDIM, 0L, DDIM,
        bo1, r16, DDIM, 0L, MROWS, DDIM, DDIM, 1.0f, FLAG_RELU);
    // 11) out = R @ Wo2t + bo2  -> f32 d_out
    wmma_gemm_seg<<<dim3(MROWS / 128, DDIM / 128, 1), dim3(256), 0, stream>>>(
        r16, DDIM, 0L, 1, MROWS,
        wt + 5L * WELEM, DDIM, 0L, DDIM,
        bo2, (float*)d_out, DDIM, 0L, MROWS, DDIM, DDIM, 1.0f, FLAG_F32);
}
